// MaskProp_19533511262834
// MI455X (gfx1250) — compile-verified, matrix operand order checked by reading, not run on hardware
//
#include <hip/hip_runtime.h>
#include <hip/hip_bf16.h>

typedef __attribute__((ext_vector_type(16))) _Float16 v16h;
typedef __attribute__((ext_vector_type(8)))  _Float16 v8h;
typedef __attribute__((ext_vector_type(8)))  float    v8f;

// ---------------------------------------------------------------------------
// Problem constants: B=8, C=256, H=W=128, WS=8 -> 2048 windows of 64 tokens.
// ---------------------------------------------------------------------------
#define NTOK   64
#define CCH    256
#define HEADS  4
#define HD     64
#define SCALE  0.125f          // hd^-0.5
#define LOWTHR 0.2f
#define BNEPS  1e-5f

// workspace byte offsets
#define WS_QW    0            // 65536 f16
#define WS_KVW   131072       // 131072 f16
#define WS_PJW   393216       // 65536 f16
#define WS_MASKR 524288       // 131072 f32 (resized mask, B x 128 x 128)
#define WS_STAT  1048576      // 512 f32 (mean, rstd per channel)

// XOR swizzle on 16B (8-half) chunks: kills LDS bank conflicts for the
// row-strided fragment loads while preserving 8-element contiguity.
__device__ __forceinline__ int swz_idx(int row, int k, int ld) {
  int c = (k >> 3) ^ (row & 7);
  return row * ld + (c << 3) + (k & 7);
}

__device__ __forceinline__ v8f wmma_f16(v16h a, v16h b, v8f c) {
  return __builtin_amdgcn_wmma_f32_16x16x32_f16(false, a, false, b,
                                                (short)0, c, false, false);
}

// A/B fragment (16x32 f16) from a row-major [rows x ld] LDS array.
// Lane l: row = row0 + l%16 ; K chunks {kb..kb+7, kb+16..kb+23}, kb = l<16?0:8.
__device__ __forceinline__ v16h ld_frag_lds(const _Float16* base, int row0, int ld, int k0) {
  int lane = threadIdx.x & 31;
  int r  = row0 + (lane & 15);
  int kb = k0 + ((lane & 16) >> 1);
  v8h lo = *(const v8h*)(base + swz_idx(r, kb, ld));
  v8h hi = *(const v8h*)(base + swz_idx(r, kb + 16, ld));
  v16h f;
#pragma unroll
  for (int i = 0; i < 8; ++i) { f[i] = lo[i]; f[i + 8] = hi[i]; }
  return f;
}

// Same fragment shape from global memory (row-major weights, no swizzle).
__device__ __forceinline__ v16h ld_frag_glb(const _Float16* base, int row0, int ld, int k0) {
  int lane = threadIdx.x & 31;
  int r  = row0 + (lane & 15);
  int kb = k0 + ((lane & 16) >> 1);
  const _Float16* p = base + r * ld + kb;
  v8h lo = *(const v8h*)(p);
  v8h hi = *(const v8h*)(p + 16);
  v16h f;
#pragma unroll
  for (int i = 0; i < 8; ++i) { f[i] = lo[i]; f[i + 8] = hi[i]; }
  return f;
}

// 16x16 f32 accumulator -> f16 LDS tile (row-major).  C/D layout: lane holds
// column n = l%16, rows m = r + (l<16 ? 0 : 8) for VGPR r.
__device__ __forceinline__ void st_tile_f16(_Float16* C, int ld, int row0, int col0, const v8f& a) {
  int lane = threadIdx.x & 31;
  int n  = col0 + (lane & 15);
  int m0 = row0 + ((lane & 16) >> 1);
#pragma unroll
  for (int r = 0; r < 8; ++r) C[swz_idx(m0 + r, n, ld)] = (_Float16)a[r];
}

// Transposed variant: element (m,n) -> C[n][m]  (used to build V^T).
__device__ __forceinline__ void st_tile_f16_tr(_Float16* C, int ld, int row0, int col0, const v8f& a) {
  int lane = threadIdx.x & 31;
  int n  = col0 + (lane & 15);
  int m0 = row0 + ((lane & 16) >> 1);
#pragma unroll
  for (int r = 0; r < 8; ++r) C[swz_idx(n, m0 + r, ld)] = (_Float16)a[r];
}

// f32 tile store with scale (attention scores; read row-wise by softmax, no swizzle).
__device__ __forceinline__ void st_tile_f32s(float* C, int ld, int row0, int col0,
                                             const v8f& a, float s) {
  int lane = threadIdx.x & 31;
  int n  = col0 + (lane & 15);
  int m0 = row0 + ((lane & 16) >> 1);
#pragma unroll
  for (int r = 0; r < 8; ++r) C[(m0 + r) * ld + n] = a[r] * s;
}

// (64 x 256) = (64 x 256 LDS) @ (256 x 256 weight)^T.
// Each wave owns ALL 4 row bands x 2 column tiles -> only 2 global B-fragment
// loads per k-step (no cross-wave duplication), software-pipelined one k-step
// ahead, and 8 back-to-back WMMAs per k-step.
__device__ __forceinline__ void gemm_act_weight(const _Float16* A, const _Float16* W,
                                                _Float16* C, bool transposeC, int wid) {
  int cb = wid * 32;                                 // this wave's 32 columns
  v8f acc[4][2] = {};
  v16h b0 = ld_frag_glb(W, cb, CCH, 0);
  v16h b1 = ld_frag_glb(W, cb + 16, CCH, 0);
#pragma unroll
  for (int ks = 0; ks < CCH; ks += 32) {
    v16h cb0 = b0, cb1 = b1;
    if (ks + 32 < CCH) {                             // prefetch next k-step
      b0 = ld_frag_glb(W, cb, CCH, ks + 32);
      b1 = ld_frag_glb(W, cb + 16, CCH, ks + 32);
    }
    v16h a0 = ld_frag_lds(A, 0,  CCH, ks);
    v16h a1 = ld_frag_lds(A, 16, CCH, ks);
    v16h a2 = ld_frag_lds(A, 32, CCH, ks);
    v16h a3 = ld_frag_lds(A, 48, CCH, ks);
    acc[0][0] = wmma_f16(a0, cb0, acc[0][0]);
    acc[0][1] = wmma_f16(a0, cb1, acc[0][1]);
    acc[1][0] = wmma_f16(a1, cb0, acc[1][0]);
    acc[1][1] = wmma_f16(a1, cb1, acc[1][1]);
    acc[2][0] = wmma_f16(a2, cb0, acc[2][0]);
    acc[2][1] = wmma_f16(a2, cb1, acc[2][1]);
    acc[3][0] = wmma_f16(a3, cb0, acc[3][0]);
    acc[3][1] = wmma_f16(a3, cb1, acc[3][1]);
  }
#pragma unroll
  for (int m = 0; m < 4; ++m)
#pragma unroll
    for (int j = 0; j < 2; ++j) {
      if (transposeC) st_tile_f16_tr(C, HD, m * 16, cb + j * 16, acc[m][j]);
      else            st_tile_f16(C, CCH, m * 16, cb + j * 16, acc[m][j]);
    }
}

// ---------------------------------------------------------------------------
// Kernel A: fp32 weights -> fp16 workspace copies.
// ---------------------------------------------------------------------------
__global__ void cvt_weights_kernel(const float* __restrict__ qw, const float* __restrict__ kvw,
                                   const float* __restrict__ pjw, _Float16* __restrict__ qh,
                                   _Float16* __restrict__ kvh, _Float16* __restrict__ pjh) {
  int i = blockIdx.x * 256 + threadIdx.x;          // 262144 total
  if (i < 65536)            qh[i]           = (_Float16)qw[i];
  else if (i < 196608)      kvh[i - 65536]  = (_Float16)kvw[i - 65536];
  else                      pjh[i - 196608] = (_Float16)pjw[i - 196608];
}

// ---------------------------------------------------------------------------
// Kernel B: bilinear (align_corners=False) 64x64 -> 128x128, times feat_mask.
// ---------------------------------------------------------------------------
__global__ void resize_mask_kernel(const float* __restrict__ mask,
                                   const float* __restrict__ fmask,
                                   float* __restrict__ maskr) {
  int idx = blockIdx.x * 256 + threadIdx.x;        // 131072 total
  int b = idx >> 14, y = (idx >> 7) & 127, x = idx & 127;
  float fy = 0.5f * y - 0.25f, fx = 0.5f * x - 0.25f;
  float y0f = floorf(fy), x0f = floorf(fx);
  float wy = fy - y0f, wx = fx - x0f;
  int y0 = min(max((int)y0f, 0), 63), y1 = min(max((int)y0f + 1, 0), 63);
  int x0 = min(max((int)x0f, 0), 63), x1 = min(max((int)x0f + 1, 0), 63);
  const float* mb = mask + b * 4096;
  float v = (1.f - wy) * ((1.f - wx) * mb[y0 * 64 + x0] + wx * mb[y0 * 64 + x1])
          +        wy  * ((1.f - wx) * mb[y1 * 64 + x0] + wx * mb[y1 * 64 + x1]);
  maskr[idx] = v * fmask[y * 128 + x];
}

// ---------------------------------------------------------------------------
// Kernel C: fully fused per-window cross attention.  1 block = 1 window.
// 160 KB dynamic LDS, regions aliased across phases:
//   [0,64K)   : X(32K)+T(32K)  ->  scores S f32(64K)   ->  Y^T f32(64K)
//   [64K,96K) : Q(32K)         ->  P f16(32K)
//   [96K,128K): K(32K)         ->  O f16(32K)
//   [128K,160K): V^T f16(32K)
// ---------------------------------------------------------------------------
__global__ __launch_bounds__(256)
void attn_window_kernel(const float* __restrict__ pre,  const float* __restrict__ pmask,
                        const float* __restrict__ feat, const float* __restrict__ fmask,
                        const _Float16* __restrict__ qw, const _Float16* __restrict__ kvw,
                        const _Float16* __restrict__ pjw, const float* __restrict__ pjb,
                        const float* __restrict__ maskr, float* __restrict__ out) {
  extern __shared__ __align__(16) char smem[];
  _Float16* sm_x  = (_Float16*)(smem);
  _Float16* sm_t  = (_Float16*)(smem + 32768);
  float*    sm_s  = (float*)(smem);                // aliases X+T (dead)
  float*    sm_y  = (float*)(smem);                // aliases S (dead)
  _Float16* sm_q  = (_Float16*)(smem + 65536);
  _Float16* sm_p  = sm_q;                          // aliases Q (dead)
  _Float16* sm_k  = (_Float16*)(smem + 98304);
  _Float16* sm_o  = sm_k;                          // aliases K (dead)
  _Float16* sm_vt = (_Float16*)(smem + 131072);
  float*    sm_m  = (float*)(smem + 65536);        // 64 mask values, pre-Q only

  int w  = blockIdx.x;
  int b  = w >> 8, wr = (w >> 4) & 15, wc = w & 15;
  int tid = threadIdx.x, wid = tid >> 5, lane = tid & 31;

  // ---- Phase 0: stage window activations as swizzled f16 --------------------
  for (int it = 0; it < 8; ++it) {
    int seg = tid + it * 256;                      // (ch, row-in-window)
    int ch = seg >> 3, i = seg & 7;
    int gi = wr * 8 + i, gx = wc * 8;
    size_t go = (((size_t)(b * CCH + ch)) << 14) + gi * 128 + gx;
    const float* fr  = feat  + go;
    const float* pr  = pre   + go;
    const float* fmr = fmask + gi * 128 + gx;
    const float* pmr = pmask + gi * 128 + gx;
    const float* mrr = maskr + ((size_t)b << 14) + gi * 128 + gx;
#pragma unroll
    for (int j = 0; j < 8; ++j) {
      int tok = i * 8 + j;
      sm_x[swz_idx(tok, ch, CCH)] = (_Float16)(fr[j] * fmr[j]);
      sm_t[swz_idx(tok, ch, CCH)] = (_Float16)(pr[j] * pmr[j] * mrr[j]);
      if (ch == 0) sm_m[tok] = mrr[j];
    }
  }
  __syncthreads();

  float mmax = 0.0f;
  for (int i = 0; i < NTOK; ++i) mmax = fmaxf(mmax, sm_m[i]);
  float keepf = (mmax >= LOWTHR) ? 1.0f : 0.0f;
  __syncthreads();

  // ---- Phase 1: Q = X Wq^T ; K = T Wk^T ; V^T from T Wv^T -------------------
  gemm_act_weight(sm_x, qw,               sm_q,  false, wid);
  gemm_act_weight(sm_t, kvw,              sm_k,  false, wid);   // kv rows 0..255
  gemm_act_weight(sm_t, kvw + 256 * CCH,  sm_vt, true,  wid);   // kv rows 256..511
  __syncthreads();

  // ---- Phase 2: per-head scores S = scale * Q_h K_h^T -----------------------
  // Wave pair per head: wave owns 2 row bands x 4 col tiles.
  {
    int head = wid >> 1, rb = (wid & 1) * 2;
    v8f acc[2][4] = {};
#pragma unroll
    for (int ks = 0; ks < HD; ks += 32) {
      v16h a0 = ld_frag_lds(sm_q, (rb + 0) * 16, CCH, head * HD + ks);
      v16h a1 = ld_frag_lds(sm_q, (rb + 1) * 16, CCH, head * HD + ks);
      v16h b0 = ld_frag_lds(sm_k, 0,  CCH, head * HD + ks);
      v16h b1 = ld_frag_lds(sm_k, 16, CCH, head * HD + ks);
      v16h b2 = ld_frag_lds(sm_k, 32, CCH, head * HD + ks);
      v16h b3 = ld_frag_lds(sm_k, 48, CCH, head * HD + ks);
      acc[0][0] = wmma_f16(a0, b0, acc[0][0]);
      acc[0][1] = wmma_f16(a0, b1, acc[0][1]);
      acc[0][2] = wmma_f16(a0, b2, acc[0][2]);
      acc[0][3] = wmma_f16(a0, b3, acc[0][3]);
      acc[1][0] = wmma_f16(a1, b0, acc[1][0]);
      acc[1][1] = wmma_f16(a1, b1, acc[1][1]);
      acc[1][2] = wmma_f16(a1, b2, acc[1][2]);
      acc[1][3] = wmma_f16(a1, b3, acc[1][3]);
    }
#pragma unroll
    for (int i = 0; i < 2; ++i)
#pragma unroll
      for (int j = 0; j < 4; ++j)
        st_tile_f32s(sm_s + head * NTOK * NTOK, NTOK, (rb + i) * 16, j * 16,
                     acc[i][j], SCALE);
  }
  __syncthreads();

  // ---- Phase 3: softmax, one row per thread (4 heads * 64 rows = 256) -------
  {
    int head = tid >> 6, row = tid & 63;
    const float* srow = sm_s + head * NTOK * NTOK + row * NTOK;
    float mx = -1e30f;
    for (int i = 0; i < NTOK; ++i) mx = fmaxf(mx, srow[i]);
    float sum = 0.f;
    for (int i = 0; i < NTOK; ++i) sum += __expf(srow[i] - mx);
    float inv = 1.0f / sum;
    _Float16* pbase = sm_p + head * NTOK * HD;
    for (int i = 0; i < NTOK; ++i)
      pbase[swz_idx(row, i, HD)] = (_Float16)(__expf(srow[i] - mx) * inv);
  }
  __syncthreads();

  // ---- Phase 4: O_h = P_h V_h  (V pre-transposed) ---------------------------
  {
    int head = wid >> 1, rb = (wid & 1) * 2;
    const _Float16* pb = sm_p + head * NTOK * HD;
    v8f acc[2][4] = {};
#pragma unroll
    for (int ks = 0; ks < NTOK; ks += 32) {
      v16h a0 = ld_frag_lds(pb, (rb + 0) * 16, HD, ks);
      v16h a1 = ld_frag_lds(pb, (rb + 1) * 16, HD, ks);
      v16h b0 = ld_frag_lds(sm_vt, head * HD + 0,  HD, ks);
      v16h b1 = ld_frag_lds(sm_vt, head * HD + 16, HD, ks);
      v16h b2 = ld_frag_lds(sm_vt, head * HD + 32, HD, ks);
      v16h b3 = ld_frag_lds(sm_vt, head * HD + 48, HD, ks);
      acc[0][0] = wmma_f16(a0, b0, acc[0][0]);
      acc[0][1] = wmma_f16(a0, b1, acc[0][1]);
      acc[0][2] = wmma_f16(a0, b2, acc[0][2]);
      acc[0][3] = wmma_f16(a0, b3, acc[0][3]);
      acc[1][0] = wmma_f16(a1, b0, acc[1][0]);
      acc[1][1] = wmma_f16(a1, b1, acc[1][1]);
      acc[1][2] = wmma_f16(a1, b2, acc[1][2]);
      acc[1][3] = wmma_f16(a1, b3, acc[1][3]);
    }
#pragma unroll
    for (int i = 0; i < 2; ++i)
#pragma unroll
      for (int j = 0; j < 4; ++j)
        st_tile_f16(sm_o, CCH, (rb + i) * 16, head * HD + j * 16, acc[i][j]);
  }
  __syncthreads();

  // ---- Phase 5: proj + bias, gate by keep, stage Y^T in LDS -----------------
  {
    int cb = wid * 32;
    v8f acc[4][2] = {};
    v16h b0 = ld_frag_glb(pjw, cb, CCH, 0);
    v16h b1 = ld_frag_glb(pjw, cb + 16, CCH, 0);
#pragma unroll
    for (int ks = 0; ks < CCH; ks += 32) {
      v16h cb0 = b0, cb1 = b1;
      if (ks + 32 < CCH) {
        b0 = ld_frag_glb(pjw, cb, CCH, ks + 32);
        b1 = ld_frag_glb(pjw, cb + 16, CCH, ks + 32);
      }
      v16h a0 = ld_frag_lds(sm_o, 0,  CCH, ks);
      v16h a1 = ld_frag_lds(sm_o, 16, CCH, ks);
      v16h a2 = ld_frag_lds(sm_o, 32, CCH, ks);
      v16h a3 = ld_frag_lds(sm_o, 48, CCH, ks);
      acc[0][0] = wmma_f16(a0, cb0, acc[0][0]);
      acc[0][1] = wmma_f16(a0, cb1, acc[0][1]);
      acc[1][0] = wmma_f16(a1, cb0, acc[1][0]);
      acc[1][1] = wmma_f16(a1, cb1, acc[1][1]);
      acc[2][0] = wmma_f16(a2, cb0, acc[2][0]);
      acc[2][1] = wmma_f16(a2, cb1, acc[2][1]);
      acc[3][0] = wmma_f16(a3, cb0, acc[3][0]);
      acc[3][1] = wmma_f16(a3, cb1, acc[3][1]);
    }
#pragma unroll
    for (int m = 0; m < 4; ++m)
#pragma unroll
      for (int j = 0; j < 2; ++j) {
        int n  = cb + j * 16 + (lane & 15);
        int m0 = m * 16 + ((lane & 16) >> 1);
        float bv = pjb[n];
#pragma unroll
        for (int r = 0; r < 8; ++r)
          sm_y[n * NTOK + m0 + r] = (acc[m][j][r] + bv) * keepf;  // Y^T[ch][tok]
      }
  }
  __syncthreads();

  // ---- Phase 6: x = heat + feat*feat_mask -> d_out (BN staging) -------------
  for (int e = tid; e < NTOK * CCH; e += 256) {
    int ch = e >> 6, tok = e & 63;
    int gi = wr * 8 + (tok >> 3), gj = wc * 8 + (tok & 7);
    size_t go = (((size_t)(b * CCH + ch)) << 14) + gi * 128 + gj;
    out[go] = sm_y[ch * NTOK + tok] + feat[go] * fmask[gi * 128 + gj];
  }
}

// ---------------------------------------------------------------------------
// Kernel D: per-channel batch mean / rstd over (B,H,W) = 131072 samples.
// ---------------------------------------------------------------------------
__global__ void bn_stats_kernel(const float* __restrict__ x, float* __restrict__ stats) {
  int c = blockIdx.x, t = threadIdx.x;
  float s = 0.f, sq = 0.f;
  for (int b = 0; b < 8; ++b) {
    const float* p = x + (((size_t)(b * CCH + c)) << 14);
    for (int i = t; i < 16384; i += 256) { float v = p[i]; s += v; sq += v * v; }
  }
  __shared__ float rs[256], rq[256];
  rs[t] = s; rq[t] = sq;
  __syncthreads();
  for (int o = 128; o > 0; o >>= 1) {
    if (t < o) { rs[t] += rs[t + o]; rq[t] += rq[t + o]; }
    __syncthreads();
  }
  if (t == 0) {
    float inv = 1.0f / 131072.0f;
    float mean = rs[0] * inv;
    float var  = rq[0] * inv - mean * mean;
    stats[2 * c]     = mean;
    stats[2 * c + 1] = rsqrtf(var + BNEPS);
  }
}

__global__ void bn_apply_kernel(float* __restrict__ x, const float* __restrict__ stats,
                                const float* __restrict__ gamma, const float* __restrict__ beta) {
  size_t idx = ((size_t)blockIdx.x * 256 + threadIdx.x) * 4;
  int c = (int)((idx >> 14) & 255);
  float mean = stats[2 * c], g = gamma[c] * stats[2 * c + 1], bb = beta[c];
  float4 v = *reinterpret_cast<float4*>(x + idx);
  v.x = (v.x - mean) * g + bb;
  v.y = (v.y - mean) * g + bb;
  v.z = (v.z - mean) * g + bb;
  v.w = (v.w - mean) * g + bb;
  *reinterpret_cast<float4*>(x + idx) = v;
}

// ---------------------------------------------------------------------------
extern "C" void kernel_launch(void* const* d_in, const int* in_sizes, int n_in,
                              void* d_out, int out_size, void* d_ws, size_t ws_size,
                              hipStream_t stream) {
  (void)in_sizes; (void)n_in; (void)out_size; (void)ws_size;
  const float* pre    = (const float*)d_in[0];
  const float* pmask  = (const float*)d_in[1];
  const float* feat   = (const float*)d_in[2];
  const float* fmask  = (const float*)d_in[3];
  const float* mask   = (const float*)d_in[4];
  const float* qw_f   = (const float*)d_in[5];
  const float* kvw_f  = (const float*)d_in[6];
  const float* pjw_f  = (const float*)d_in[7];
  const float* pjb    = (const float*)d_in[8];
  const float* gamma  = (const float*)d_in[9];
  const float* beta   = (const float*)d_in[10];
  float* out = (float*)d_out;

  char* ws = (char*)d_ws;
  _Float16* qh    = (_Float16*)(ws + WS_QW);
  _Float16* kvh   = (_Float16*)(ws + WS_KVW);
  _Float16* pjh   = (_Float16*)(ws + WS_PJW);
  float*    maskr = (float*)(ws + WS_MASKR);
  float*    stats = (float*)(ws + WS_STAT);

  cvt_weights_kernel<<<1024, 256, 0, stream>>>(qw_f, kvw_f, pjw_f, qh, kvh, pjh);
  resize_mask_kernel<<<512, 256, 0, stream>>>(mask, fmask, maskr);
  attn_window_kernel<<<2048, 256, 160 * 1024, stream>>>(
      pre, pmask, feat, fmask, qh, kvh, pjh, pjb, maskr, out);
  bn_stats_kernel<<<256, 256, 0, stream>>>(out, stats);
  bn_apply_kernel<<<32768, 256, 0, stream>>>(out, stats, gamma, beta);
}
